// HouseholdODEFunc_37692632989903
// MI455X (gfx1250) — matrix operand order repaired and checked by reading, not compiled, not run on hardware
//
#include <hip/hip_runtime.h>
#include <hip/hip_bf16.h>

// ---------------------------------------------------------------------------
// MI455X (gfx1250, wave32) implementation.
// Dominant cost: 8192x8192 leave-one-out attention (~34 GFLOP) -> flash
// attention with v_wmma_f32_16x16x32_bf16 (fp32 accumulation).
// All other GEMMs go through one generic WMMA GEMM kernel.
// ---------------------------------------------------------------------------

#define N_PPL 8192
#define HDIM  128

typedef __attribute__((ext_vector_type(16))) __bf16 v16bf;
typedef __attribute__((ext_vector_type(8)))  float  v8f;

union FragAB {
    v16bf v;
    uint4 u[2];   // u[0] = elements 0..7, u[1] = elements 8..15
};

// ---------------------------------------------------------------------------
// fp32 -> bf16 elementwise convert
// ---------------------------------------------------------------------------
__global__ __launch_bounds__(256)
void cvt_f2b(const float* __restrict__ s, __bf16* __restrict__ d, int n) {
    int i = blockIdx.x * 256 + threadIdx.x;
    if (i < n) d[i] = (__bf16)s[i];
}

// zebT[j][z] = gnn_zone_embeds[z][j], padded to z<128 with zeros (64 x 128)
__global__ __launch_bounds__(256)
void build_zebT(const float* __restrict__ Z, __bf16* __restrict__ zt) {
    int i = blockIdx.x * 256 + threadIdx.x;
    if (i < 64 * 128) {
        int j = i >> 7, z = i & 127;
        zt[i] = (z < 100) ? (__bf16)Z[z * 64 + j] : (__bf16)0.0f;
    }
}

// bias_eff[o] = b_ih[o] + sum_j W_ih[o, 192+j] * (t*W_time[j] + b_time[j])
// (time_context is identical for every person -> fold into GRU input bias)
__global__ __launch_bounds__(128)
void build_bias_eff(const float* __restrict__ t, const float* __restrict__ Wt,
                    const float* __restrict__ bt, const float* __restrict__ Wih,
                    const float* __restrict__ bih, float* __restrict__ be) {
    int o = blockIdx.x * 128 + threadIdx.x;
    if (o < 384) {
        float tv = t[0];
        float s = bih[o];
        for (int j = 0; j < 32; ++j)
            s += Wih[o * 224 + 192 + j] * (tv * Wt[j] + bt[j]);
        be[o] = s;
    }
}

// ---------------------------------------------------------------------------
// Generic WMMA GEMM:  C[M=8192, Nout] = A[M, K] * W[Nout, K]^T (+bias)(+Cacc)
// A, W are bf16.  K = 32*Kchunks.  grid = (8192/64, Ntiles), block = 128 (4 waves).
// Each wave: 16 rows x 16 cols, K-loop of v_wmma_f32_16x16x32_bf16.
// Out-of-bounds output columns (nbound not multiple of 16): W row index is
// clamped for the loads (no divergence / zero-fill in the K-loop); those
// lanes compute garbage that is simply never stored.
// Stores: fp32 (outF) and/or bf16 (outB); transB -> outB[col*ldc + row].
// ---------------------------------------------------------------------------
__global__ __launch_bounds__(128)
void gemm_wmma(const __bf16* __restrict__ A, int lda,
               const __bf16* __restrict__ W, int ldw, int wcoloff,
               const float* __restrict__ bias,
               const float* __restrict__ Cacc,
               float* __restrict__ outF, __bf16* __restrict__ outB,
               int ldc, int nbound, int Kchunks, int transB) {
    const int wave = threadIdx.x >> 5;
    const int lane = threadIdx.x & 31;
    const int n16  = lane & 15;
    const int half = lane >> 4;
    const int rowbase = blockIdx.x * 64 + wave * 16;
    const int col     = blockIdx.y * 16 + n16;
    const int arow    = rowbase + n16;
    const bool wvalid = (col < nbound);
    const int  wrow   = wvalid ? col : (nbound - 1);   // clamped load row

    const __bf16* ap0 = A + (size_t)arow * lda + 8 * half;
    const __bf16* wp0 = W + (size_t)wrow * ldw + wcoloff + 8 * half;

    v8f acc = {};
#pragma unroll 4
    for (int c = 0; c < Kchunks; ++c) {
        FragAB a, b;
        // A fragment (ISA layout): row = lane%16, K = e + 8*half (e<8), 16+(e-8)+8*half (e>=8)
        const __bf16* ap = ap0 + c * 32;
        a.u[0] = *(const uint4*)ap;
        a.u[1] = *(const uint4*)(ap + 16);
        // B fragment: col = lane%16, K = e + 8*half (e<8), 16+(e-8)+8*half (e>=8)
        const __bf16* wp = wp0 + c * 32;
        b.u[0] = *(const uint4*)wp;
        b.u[1] = *(const uint4*)(wp + 16);
        acc = __builtin_amdgcn_wmma_f32_16x16x32_bf16(
            false, a.v, false, b.v, (short)0, acc, false, false);
    }

    if (wvalid) {
        const float bval = (bias != nullptr) ? bias[col] : 0.0f;
#pragma unroll
        for (int r = 0; r < 8; ++r) {
            const int row = rowbase + r + 8 * half;   // C layout: row = r + 8*half
            float val = acc[r] + bval;
            if (Cacc) val += Cacc[(size_t)row * ldc + col];
            if (outF) outF[(size_t)row * ldc + col] = val;
            if (outB) {
                if (transB) outB[(size_t)col * ldc + row] = (__bf16)val;
                else        outB[(size_t)row * ldc + col] = (__bf16)val;
            }
        }
    }
}

// ---------------------------------------------------------------------------
// Zone softmax: one wave per row over 100 logits; writes bf16 probs padded
// to 128 columns (cols >= 100 are zero) so spatial GEMM uses K=128.
// ---------------------------------------------------------------------------
__global__ __launch_bounds__(256)
void zone_softmax(const float* __restrict__ locf, __bf16* __restrict__ probsb) {
    const int wave = threadIdx.x >> 5, lane = threadIdx.x & 31;
    const int row = blockIdx.x * 8 + wave;
    const float* src = locf + (size_t)row * 100;
    float v[4];
    float mx = -1e30f;
#pragma unroll
    for (int i = 0; i < 4; ++i) {
        int c = lane + 32 * i;
        v[i] = (c < 100) ? src[c] : -1e30f;
        mx = fmaxf(mx, v[i]);
    }
    for (int m = 1; m < 32; m <<= 1) mx = fmaxf(mx, __shfl_xor(mx, m, 32));
    float s = 0.0f;
#pragma unroll
    for (int i = 0; i < 4; ++i) {
        int c = lane + 32 * i;
        if (c < 100) { v[i] = __expf(v[i] - mx); s += v[i]; } else v[i] = 0.0f;
    }
    for (int m = 1; m < 32; m <<= 1) s += __shfl_xor(s, m, 32);
    const float inv = 1.0f / s;
    __bf16* dst = probsb + (size_t)row * 128;
#pragma unroll
    for (int i = 0; i < 4; ++i) { int c = lane + 32 * i; dst[c] = (__bf16)(v[i] * inv); }
}

// ---------------------------------------------------------------------------
// Flash attention with diagonal mask.
// Block = 128 threads (4 waves) -> 64 query rows; stream 64-key tiles.
// Scores sigma ~= 0.33 -> plain exp() online softmax (no running max needed).
// Per key tile per wave: 16 WMMA (QK^T) + 16 WMMA (P*V).
// V is pre-transposed in global (vbT[128][8192]) so both B-fragment loads
// come from row-major LDS as aligned b128 reads.
// ---------------------------------------------------------------------------
__global__ __launch_bounds__(128)
void flash_attn(const __bf16* __restrict__ qb, const __bf16* __restrict__ kb,
                const __bf16* __restrict__ vbT, __bf16* __restrict__ socialb) {
    __shared__ __bf16 Kt[64][136];    // [key][dim], padded
    __shared__ __bf16 Vt[128][72];    // [dim][key], padded (from vbT, row-major copy)
    __shared__ __bf16 Ps[4][16][72];  // per-wave P staging [row][key], padded

    const int wave = threadIdx.x >> 5, lane = threadIdx.x & 31;
    const int n16 = lane & 15, half = lane >> 4;
    const int rowbase = blockIdx.x * 64;
    const int qrow = rowbase + wave * 16 + n16;

    // Q fragments for this wave's 16 rows (K = 128 -> 4 chunks), loaded once.
    FragAB qf[4];
#pragma unroll
    for (int c = 0; c < 4; ++c) {
        const __bf16* qp = qb + (size_t)qrow * HDIM + c * 32 + 8 * half;
        qf[c].u[0] = *(const uint4*)qp;
        qf[c].u[1] = *(const uint4*)(qp + 16);
    }

    v8f acc[8];
#pragma unroll
    for (int j = 0; j < 8; ++j) acc[j] = (v8f){};
    float psum[8];
#pragma unroll
    for (int r = 0; r < 8; ++r) psum[r] = 0.0f;
    const float scale = 0.08838834764831845f;   // 1/sqrt(128)

    for (int kbs = 0; kbs < N_PPL; kbs += 64) {
        // Cooperative tile loads (coalesced b128).
        {
            const int key = threadIdx.x >> 1, dseg = (threadIdx.x & 1) * 64;
            const uint4* ks = (const uint4*)(kb + (size_t)(kbs + key) * HDIM + dseg);
            uint4* kd = (uint4*)&Kt[key][dseg];
#pragma unroll
            for (int i = 0; i < 8; ++i) kd[i] = ks[i];
            const uint4* vs = (const uint4*)(vbT + (size_t)threadIdx.x * N_PPL + kbs);
            uint4* vd = (uint4*)&Vt[threadIdx.x][0];
#pragma unroll
            for (int i = 0; i < 8; ++i) vd[i] = vs[i];
            if (kbs + 64 < N_PPL)  // prefetch next K tile -> global_prefetch_b8
                __builtin_prefetch(kb + (size_t)(kbs + 64 + key) * HDIM + dseg, 0, 3);
        }
        __syncthreads();

        // S = Q K^T over 4 key sub-tiles, then P = exp(S*scale) with diag mask.
#pragma unroll
        for (int sub = 0; sub < 4; ++sub) {
            v8f s = {};
#pragma unroll
            for (int c = 0; c < 4; ++c) {
                FragAB b;
                const __bf16* kp = &Kt[sub * 16 + n16][c * 32 + 8 * half];
                b.u[0] = *(const uint4*)kp;
                b.u[1] = *(const uint4*)(kp + 16);
                s = __builtin_amdgcn_wmma_f32_16x16x32_bf16(
                    false, qf[c].v, false, b.v, (short)0, s, false, false);
            }
            const int colg = kbs + sub * 16 + n16;
#pragma unroll
            for (int r = 0; r < 8; ++r) {
                const int rowg = rowbase + wave * 16 + r + 8 * half;
                const float p = (rowg == colg) ? 0.0f : __expf(s[r] * scale);
                psum[r] += p;
                Ps[wave][r + 8 * half][sub * 16 + n16] = (__bf16)p;
            }
        }
        // Intra-wave LDS RAW (cross-lane): wait for DS stores to retire.
        asm volatile("s_wait_dscnt 0x0" ::: "memory");

        // O += P * V : A = P (16x64 -> 2 K-chunks), B = V^T rows (output dims).
#pragma unroll
        for (int jt = 0; jt < 8; ++jt) {
#pragma unroll
            for (int c2 = 0; c2 < 2; ++c2) {
                FragAB pa, vb;
                const __bf16* pp = &Ps[wave][n16][c2 * 32 + 8 * half];
                pa.u[0] = *(const uint4*)pp;
                pa.u[1] = *(const uint4*)(pp + 16);
                const __bf16* vp = &Vt[jt * 16 + n16][c2 * 32 + 8 * half];
                vb.u[0] = *(const uint4*)vp;
                vb.u[1] = *(const uint4*)(vp + 16);
                acc[jt] = __builtin_amdgcn_wmma_f32_16x16x32_bf16(
                    false, pa.v, false, vb.v, (short)0, acc[jt], false, false);
            }
        }
        __syncthreads();
    }

    // Row sums: reduce per-lane partials across the 16 lanes of each half.
    float tot[8];
#pragma unroll
    for (int r = 0; r < 8; ++r) {
        float s = psum[r];
        for (int m = 1; m < 16; m <<= 1) s += __shfl_xor(s, m, 16);
        tot[r] = s;
    }
#pragma unroll
    for (int jt = 0; jt < 8; ++jt) {
#pragma unroll
        for (int r = 0; r < 8; ++r) {
            const int rowg = rowbase + wave * 16 + r + 8 * half;
            socialb[(size_t)rowg * HDIM + jt * 16 + n16] = (__bf16)(acc[jt][r] / tot[r]);
        }
    }
}

// ---------------------------------------------------------------------------
// Fused GRU cell + LayerNorm(d). One wave per row (4 cols/lane).
// ---------------------------------------------------------------------------
__global__ __launch_bounds__(256)
void gru_ln(const float* __restrict__ gx, const float* __restrict__ gh,
            const float* __restrict__ h0, const float* __restrict__ gamma,
            const float* __restrict__ beta, float* __restrict__ out) {
    const int wave = threadIdx.x >> 5, lane = threadIdx.x & 31;
    const int row = blockIdx.x * 8 + wave;
    const float* gxr = gx + (size_t)row * 384;
    const float* ghr = gh + (size_t)row * 384;
    const float* hp  = h0 + (size_t)row * 128;
    float d[4];
    float s1 = 0.0f, s2 = 0.0f;
#pragma unroll
    for (int i = 0; i < 4; ++i) {
        const int j = lane + 32 * i;
        const float xr = gxr[j], xz = gxr[j + 128], xn = gxr[j + 256];
        const float hr = ghr[j], hz = ghr[j + 128], hn = ghr[j + 256];
        const float h = hp[j];
        const float r = 1.0f / (1.0f + __expf(-(xr + hr)));
        const float z = 1.0f / (1.0f + __expf(-(xz + hz)));
        const float ng = tanhf(xn + r * hn);
        const float hnew = (1.0f - z) * ng + z * h;
        d[i] = hnew - h;
        s1 += d[i];
        s2 += d[i] * d[i];
    }
    for (int m = 1; m < 32; m <<= 1) {
        s1 += __shfl_xor(s1, m, 32);
        s2 += __shfl_xor(s2, m, 32);
    }
    const float mu  = s1 * (1.0f / 128.0f);
    const float var = s2 * (1.0f / 128.0f) - mu * mu;
    const float rs  = rsqrtf(var + 1e-5f);
    float* o = out + (size_t)row * 128;
#pragma unroll
    for (int i = 0; i < 4; ++i) {
        const int j = lane + 32 * i;
        o[j] = (d[i] - mu) * rs * gamma[j] + beta[j];
    }
}

// ---------------------------------------------------------------------------
// Host launcher
// ---------------------------------------------------------------------------
extern "C" void kernel_launch(void* const* d_in, const int* in_sizes, int n_in,
                              void* d_out, int out_size, void* d_ws, size_t ws_size,
                              hipStream_t stream) {
    (void)in_sizes; (void)n_in; (void)out_size; (void)ws_size;
    const float* t        = (const float*)d_in[0];
    const float* person_h = (const float*)d_in[1];
    const float* zemb     = (const float*)d_in[2];
    const float* W_time   = (const float*)d_in[3];
    const float* b_time   = (const float*)d_in[4];
    const float* W_loc    = (const float*)d_in[5];
    const float* b_loc    = (const float*)d_in[6];
    const float* Wq       = (const float*)d_in[7];
    const float* bq       = (const float*)d_in[8];
    const float* Wk       = (const float*)d_in[9];
    const float* bk       = (const float*)d_in[10];
    const float* Wv       = (const float*)d_in[11];
    const float* bv       = (const float*)d_in[12];
    const float* W_merge  = (const float*)d_in[13];
    const float* b_merge  = (const float*)d_in[14];
    const float* W_ih     = (const float*)d_in[15];
    const float* b_ih     = (const float*)d_in[16];
    const float* W_hh     = (const float*)d_in[17];
    const float* b_hh     = (const float*)d_in[18];
    const float* gamma    = (const float*)d_in[19];
    const float* beta     = (const float*)d_in[20];
    float* out = (float*)d_out;

    // ---- workspace carve (256B aligned) ----
    uint8_t* p = (uint8_t*)d_ws;
    auto carve = [&](size_t bytes) -> void* {
        void* r = (void*)p;
        p += (bytes + 255) & ~(size_t)255;
        return r;
    };
    __bf16* hb       = (__bf16*)carve((size_t)N_PPL * 128 * 2);
    __bf16* qb       = (__bf16*)carve((size_t)N_PPL * 128 * 2);
    __bf16* kb       = (__bf16*)carve((size_t)N_PPL * 128 * 2);
    __bf16* vbT      = (__bf16*)carve((size_t)N_PPL * 128 * 2);  // [128][8192]
    __bf16* probsb   = (__bf16*)carve((size_t)N_PPL * 128 * 2);  // padded to 128
    __bf16* spatialb = (__bf16*)carve((size_t)N_PPL * 64 * 2);
    __bf16* socialb  = (__bf16*)carve((size_t)N_PPL * 128 * 2);
    __bf16* hintb    = (__bf16*)carve((size_t)N_PPL * 128 * 2);
    float*  locf     = (float*)carve((size_t)N_PPL * 100 * 4);
    float*  hintf    = (float*)carve((size_t)N_PPL * 128 * 4);
    float*  gxf      = (float*)carve((size_t)N_PPL * 384 * 4);
    float*  ghf      = (float*)carve((size_t)N_PPL * 384 * 4);
    __bf16* wqb      = (__bf16*)carve(128 * 128 * 2);
    __bf16* wkb      = (__bf16*)carve(128 * 128 * 2);
    __bf16* wvb      = (__bf16*)carve(128 * 128 * 2);
    __bf16* wlocb    = (__bf16*)carve(100 * 128 * 2);
    __bf16* wmergeb  = (__bf16*)carve(128 * 256 * 2);
    __bf16* wihb     = (__bf16*)carve(384 * 224 * 2);
    __bf16* whhb     = (__bf16*)carve(384 * 128 * 2);
    __bf16* zebT     = (__bf16*)carve(64 * 128 * 2);
    float*  bias_eff = (float*)carve(384 * 4);

    auto cvt = [&](const float* s, __bf16* d, int n) {
        cvt_f2b<<<(n + 255) / 256, 256, 0, stream>>>(s, d, n);
    };
    cvt(person_h, hb, N_PPL * 128);
    cvt(Wq, wqb, 128 * 128);
    cvt(Wk, wkb, 128 * 128);
    cvt(Wv, wvb, 128 * 128);
    cvt(W_loc, wlocb, 100 * 128);
    cvt(W_merge, wmergeb, 128 * 256);
    cvt(W_ih, wihb, 384 * 224);
    cvt(W_hh, whhb, 384 * 128);
    build_zebT<<<(64 * 128 + 255) / 256, 256, 0, stream>>>(zemb, zebT);
    build_bias_eff<<<3, 128, 0, stream>>>(t, W_time, b_time, W_ih, b_ih, bias_eff);

    const dim3 blk(128);
    // q, k (bf16), v (bf16, transposed store)
    gemm_wmma<<<dim3(128, 8), blk, 0, stream>>>(hb, 128, wqb, 128, 0, bq, nullptr,
                                                nullptr, qb, 128, 128, 4, 0);
    gemm_wmma<<<dim3(128, 8), blk, 0, stream>>>(hb, 128, wkb, 128, 0, bk, nullptr,
                                                nullptr, kb, 128, 128, 4, 0);
    gemm_wmma<<<dim3(128, 8), blk, 0, stream>>>(hb, 128, wvb, 128, 0, bv, nullptr,
                                                nullptr, vbT, N_PPL, 128, 4, 1);
    // zone logits (Nout=100) -> softmax -> spatial = probs @ Z
    gemm_wmma<<<dim3(128, 7), blk, 0, stream>>>(hb, 128, wlocb, 128, 0, b_loc, nullptr,
                                                locf, nullptr, 100, 100, 4, 0);
    zone_softmax<<<N_PPL / 8, 256, 0, stream>>>(locf, probsb);
    gemm_wmma<<<dim3(128, 4), blk, 0, stream>>>(probsb, 128, zebT, 128, 0, nullptr,
                                                nullptr, nullptr, spatialb, 64, 64, 4, 0);
    // leave-one-out attention
    flash_attn<<<N_PPL / 64, 128, 0, stream>>>(qb, kb, vbT, socialb);
    // h_interacted = [h, social] @ W_merge^T + b_merge  (two K-segments)
    gemm_wmma<<<dim3(128, 8), blk, 0, stream>>>(hb, 128, wmergeb, 256, 0, b_merge,
                                                nullptr, hintf, nullptr, 128, 128, 4, 0);
    gemm_wmma<<<dim3(128, 8), blk, 0, stream>>>(socialb, 128, wmergeb, 256, 128, nullptr,
                                                hintf, nullptr, hintb, 128, 128, 4, 0);
    // gx = gru_input @ W_ih^T + bias_eff  (h_interacted seg + spatial seg; time folded)
    gemm_wmma<<<dim3(128, 24), blk, 0, stream>>>(hintb, 128, wihb, 224, 0, bias_eff,
                                                 nullptr, gxf, nullptr, 384, 384, 4, 0);
    gemm_wmma<<<dim3(128, 24), blk, 0, stream>>>(spatialb, 64, wihb, 224, 128, nullptr,
                                                 gxf, gxf, nullptr, 384, 384, 2, 0);
    // gh = h @ W_hh^T + b_hh
    gemm_wmma<<<dim3(128, 24), blk, 0, stream>>>(hb, 128, whhb, 128, 0, b_hh, nullptr,
                                                 ghf, nullptr, 384, 384, 4, 0);
    // GRU + LayerNorm
    gru_ln<<<N_PPL / 8, 256, 0, stream>>>(gxf, ghf, person_h, gamma, beta, out);
}
